// Attention_86131274154446
// MI455X (gfx1250) — matrix operand order repaired and checked by reading, not compiled, not run on hardware
//
#include <hip/hip_runtime.h>
#include <hip/hip_bf16.h>
#include <math.h>

typedef __bf16 bf16_t;
typedef __attribute__((ext_vector_type(16))) __bf16 v16bf;
typedef __attribute__((ext_vector_type(8)))  __bf16 v8bf;
typedef __attribute__((ext_vector_type(8)))  float  v8f;

#define NEG_INF (-3.402823466e38f)
#define EPSV 1e-6f

#if defined(__AMDGCN__) && __has_builtin(__builtin_amdgcn_global_load_async_to_lds_b128)
#define USE_ASYNC_LDS 1
typedef __attribute__((vector_size(16))) int vsi4;
typedef __attribute__((address_space(1))) vsi4 as1_vsi4;
typedef __attribute__((address_space(3))) vsi4 as3_vsi4;
#define AS1P(p) ((as1_vsi4*)(p))
#define AS3P(p) ((as3_vsi4*)(p))
static __device__ __forceinline__ void wait_async0() {
#if __has_builtin(__builtin_amdgcn_s_wait_asynccnt)
    __builtin_amdgcn_s_wait_asynccnt(0);
#else
    asm volatile("s_wait_asynccnt 0x0" ::: "memory");
#endif
}
#else
#define USE_ASYNC_LDS 0
#endif

// ---------------------------------------------------------------------------
// f32 -> bf16 convert (flat)
// ---------------------------------------------------------------------------
__global__ void f2bf_kernel(const float* __restrict__ src, bf16_t* __restrict__ dst, long long n) {
    long long i = (long long)blockIdx.x * 256 + threadIdx.x;
    if (i < n) dst[i] = (bf16_t)src[i];
}

// f32 [R][C] -> bf16 transposed [C][R].  R, C multiples of 32.
__global__ __launch_bounds__(256)
void f2bf_T_kernel(const float* __restrict__ src, bf16_t* __restrict__ dst, int R, int C)
{
    __shared__ float tile[32][33];
    int c0 = blockIdx.x * 32, r0 = blockIdx.y * 32;
    int tx = threadIdx.x & 31, ty = threadIdx.x >> 5;          // ty: 0..7
    for (int i = 0; i < 32; i += 8)
        tile[ty + i][tx] = src[(long long)(r0 + ty + i) * C + c0 + tx];
    __syncthreads();
    for (int i = 0; i < 32; i += 8)
        dst[(long long)(c0 + ty + i) * R + r0 + tx] = (bf16_t)tile[tx][ty + i];
}

__global__ void zero_f32_kernel(float* __restrict__ p, long long n) {
    long long i = (long long)blockIdx.x * 256 + threadIdx.x;
    if (i < n) p[i] = 0.0f;
}

// ---------------------------------------------------------------------------
// Tiled WMMA GEMM: C[M,N] = A[M,K](bf16 row-major) * BT[N,K](bf16 row-major)^T
// (B is supplied pre-transposed so both operands stage with b128 copies).
// 128x128 tile, K-step 32, 256 threads = 8 wave32 waves (2 M x 4 N),
// each wave: 4x2 grid of 16x16 tiles -> v_wmma_f32_16x16x32_bf16.
// Requires K % 32 == 0, N % 16 == 0.  M, N guarded.
// ---------------------------------------------------------------------------
__global__ __launch_bounds__(256)
void gemm_bf16_wmma(const bf16_t* __restrict__ Ag, const bf16_t* __restrict__ BTg,
                    float* __restrict__ Cg, const float* __restrict__ bias,
                    int M, int N, int K,
                    long long sA, long long sB, long long sC)
{
    __shared__ bf16_t As[128][40];   // [m][k]  stride 80B (16B aligned)
    __shared__ bf16_t Bs[128][40];   // [n][k]

    const int tid  = threadIdx.x;
    const int lane = tid & 31;
    const int wave = tid >> 5;
    const int wm   = wave & 1;       // 0..1  (64-row slab)
    const int wn   = wave >> 1;      // 0..3  (32-col slab)
    const int m0   = blockIdx.y * 128;
    const int n0   = blockIdx.x * 128;

    const bf16_t* A  = Ag  + (long long)blockIdx.z * sA;
    const bf16_t* BT = BTg + (long long)blockIdx.z * sB;
    float*        C  = Cg  + (long long)blockIdx.z * sC;

    v8f acc[4][2];
    for (int mi = 0; mi < 4; ++mi)
        for (int ni = 0; ni < 2; ++ni)
            for (int e = 0; e < 8; ++e) acc[mi][ni][e] = 0.0f;

    const int r  = lane & 15;
    const int hi = lane >> 4;

    const int row = tid >> 1, seg = tid & 1;   // staging assignment (shared by A/B)
    const int ga  = m0 + row;                   // A source row
    const int gb  = n0 + row;                   // BT source row

    for (int k0 = 0; k0 < K; k0 += 32) {
        const bf16_t* gpA = A  + (long long)ga * K + k0 + seg * 16;
        const bf16_t* gpB = BT + (long long)gb * K + k0 + seg * 16;
        bf16_t* lpA = &As[row][seg * 16];
        bf16_t* lpB = &Bs[row][seg * 16];

        // prefetch next K tile
        if (k0 + 32 < K) {
            if (ga < M) __builtin_prefetch(gpA + 32, 0, 3);
            if (gb < N) __builtin_prefetch(gpB + 32, 0, 3);
        }

#if USE_ASYNC_LDS
        if (ga < M) {
            __builtin_amdgcn_global_load_async_to_lds_b128(AS1P(gpA),     AS3P(lpA),     0, 0);
            __builtin_amdgcn_global_load_async_to_lds_b128(AS1P(gpA + 8), AS3P(lpA + 8), 0, 0);
        } else {
            *(uint4*)lpA       = make_uint4(0,0,0,0);
            *(uint4*)(lpA + 8) = make_uint4(0,0,0,0);
        }
        if (gb < N) {
            __builtin_amdgcn_global_load_async_to_lds_b128(AS1P(gpB),     AS3P(lpB),     0, 0);
            __builtin_amdgcn_global_load_async_to_lds_b128(AS1P(gpB + 8), AS3P(lpB + 8), 0, 0);
        } else {
            *(uint4*)lpB       = make_uint4(0,0,0,0);
            *(uint4*)(lpB + 8) = make_uint4(0,0,0,0);
        }
        wait_async0();
#else
        {
            uint4 d0 = make_uint4(0,0,0,0), d1 = make_uint4(0,0,0,0);
            if (ga < M) { const uint4* p = (const uint4*)gpA; d0 = p[0]; d1 = p[1]; }
            *(uint4*)lpA       = d0;
            *(uint4*)(lpA + 8) = d1;
        }
        {
            uint4 d0 = make_uint4(0,0,0,0), d1 = make_uint4(0,0,0,0);
            if (gb < N) { const uint4* p = (const uint4*)gpB; d0 = p[0]; d1 = p[1]; }
            *(uint4*)lpB       = d0;
            *(uint4*)(lpB + 8) = d1;
        }
#endif
        __syncthreads();

        // ---- fragment loads per ISA 7.12.2 layouts ----
        v16bf af[4];
        for (int mi = 0; mi < 4; ++mi) {
            int mr = wm * 64 + mi * 16 + r;
            v8bf lo = *(const v8bf*)&As[mr][hi * 8];        // K = hi*8 .. +7
            v8bf hp = *(const v8bf*)&As[mr][16 + hi * 8];   // K = 16+hi*8 .. +7
            for (int e = 0; e < 8; ++e) { af[mi][e] = lo[e]; af[mi][8 + e] = hp[e]; }
        }
        v16bf bfr[2];
        for (int ni = 0; ni < 2; ++ni) {
            int nc = wn * 32 + ni * 16 + r;
            v8bf lo = *(const v8bf*)&Bs[nc][hi * 8];        // K = hi*8 .. +7
            v8bf hp = *(const v8bf*)&Bs[nc][16 + hi * 8];   // K = 16+hi*8 .. +7
            for (int e = 0; e < 8; ++e) { bfr[ni][e] = lo[e]; bfr[ni][8 + e] = hp[e]; }
        }

        for (int mi = 0; mi < 4; ++mi)
            for (int ni = 0; ni < 2; ++ni)
                acc[mi][ni] = __builtin_amdgcn_wmma_f32_16x16x32_bf16(
                    false, af[mi], false, bfr[ni], (short)0, acc[mi][ni], false, false);

        __syncthreads();
    }

    // ---- epilogue: C/D layout row = e + hi*8, col = r ----
    for (int mi = 0; mi < 4; ++mi)
        for (int ni = 0; ni < 2; ++ni)
            for (int e = 0; e < 8; ++e) {
                int rw = m0 + wm * 64 + mi * 16 + hi * 8 + e;
                int cl = n0 + wn * 32 + ni * 16 + r;
                if (rw < M && cl < N) {
                    float v = acc[mi][ni][e];
                    if (bias) v += bias[cl];
                    C[(long long)rw * N + cl] = v;
                }
            }
}

// ---------------------------------------------------------------------------
// Row-0 attention scoring: per (b,h) softmax(q0 . K) then sum_h attn*||v||
// ---------------------------------------------------------------------------
__global__ __launch_bounds__(256)
void score_kernel(const float* __restrict__ qkv, float* __restrict__ cls_score)
{
    const int bh = blockIdx.x, b = bh >> 4, h = bh & 15;
    const int tid = threadIdx.x;
    __shared__ float q0[64];
    __shared__ float dots[512];
    __shared__ float red[256];
    const long long base = (long long)b * 257 * 3072;

    if (tid < 64) q0[tid] = qkv[base + h * 64 + tid];
    dots[tid] = NEG_INF; dots[tid + 256] = NEG_INF;
    __syncthreads();

    float vnorm[2] = {0.f, 0.f};
    for (int it = 0; it < 2; ++it) {
        int j = tid + it * 256;
        if (j < 257) {
            const float* kr = qkv + base + (long long)j * 3072 + 1024 + h * 64;
            const float* vr = qkv + base + (long long)j * 3072 + 2048 + h * 64;
            float d = 0.f, nv = 0.f;
            for (int e = 0; e < 64; ++e) { d += q0[e] * kr[e]; float vv = vr[e]; nv += vv * vv; }
            dots[j] = d * 0.125f;           // dh^-0.5
            vnorm[it] = sqrtf(nv);
        }
    }
    __syncthreads();

    red[tid] = fmaxf(dots[tid], dots[tid + 256]); __syncthreads();
    for (int s = 128; s > 0; s >>= 1) { if (tid < s) red[tid] = fmaxf(red[tid], red[tid + s]); __syncthreads(); }
    float m = red[0]; __syncthreads();

    float p0 = (tid < 257) ? expf(dots[tid] - m) : 0.f;
    float p1 = (tid + 256 < 257) ? expf(dots[tid + 256] - m) : 0.f;
    red[tid] = p0 + p1; __syncthreads();
    for (int s = 128; s > 0; s >>= 1) { if (tid < s) red[tid] += red[tid + s]; __syncthreads(); }
    float denom = red[0];

    for (int it = 0; it < 2; ++it) {
        int j = tid + it * 256;
        if (j >= 1 && j < 257)
            atomicAdd(&cls_score[b * 256 + (j - 1)], (expf(dots[j] - m) / denom) * vnorm[it]);
    }
}

// ---------------------------------------------------------------------------
__global__ void logits_kernel(const float* __restrict__ cls_score,
                              const unsigned char* __restrict__ mask,
                              float* __restrict__ plog)
{
    int b = blockIdx.x, tid = threadIdx.x;   // 256 threads
    __shared__ float red[256];
    float v = cls_score[b * 256 + tid];
    red[tid] = v; __syncthreads();
    for (int s = 128; s > 0; s >>= 1) { if (tid < s) red[tid] += red[tid + s]; __syncthreads(); }
    float total = red[0] + EPSV;
    float pl = logf(v / total + EPSV);
    if (!mask[b * 257 + 1 + tid]) pl = NEG_INF * 0.5f;
    plog[b * 256 + tid] = pl;
}

// ---------------------------------------------------------------------------
__global__ void gumbel_argmax_kernel(const float* __restrict__ plog,
                                     const float* __restrict__ gu,
                                     int* __restrict__ sampled)
{
    int row = blockIdx.x, b = row >> 7, tid = threadIdx.x;
    __shared__ float rv[256]; __shared__ int ri[256];
    float u = gu[(long long)row * 256 + tid];
    float g = -logf(-logf(u + EPSV) + EPSV);
    rv[tid] = plog[b * 256 + tid] + g; ri[tid] = tid;
    __syncthreads();
    for (int s = 128; s > 0; s >>= 1) {
        if (tid < s) {
            if (rv[tid + s] > rv[tid] || (rv[tid + s] == rv[tid] && ri[tid + s] < ri[tid])) {
                rv[tid] = rv[tid + s]; ri[tid] = ri[tid + s];
            }
        }
        __syncthreads();
    }
    if (tid == 0) sampled[row] = ri[0] + 1;
}

// ---------------------------------------------------------------------------
__device__ inline void bitonic128(int* s, int tid) {
    for (int k = 2; k <= 128; k <<= 1)
        for (int j = k >> 1; j > 0; j >>= 1) {
            int ixj = tid ^ j;
            if (ixj > tid) {
                int a = s[tid], c = s[ixj];
                bool up = ((tid & k) == 0);
                if ((a > c) == up) { s[tid] = c; s[ixj] = a; }
            }
            __syncthreads();
        }
}

__global__ __launch_bounds__(128)
void ids_kernel(const int* __restrict__ sampled, int* __restrict__ ids_int,
                float* __restrict__ out_mask, float* __restrict__ out_ids)
{
    int b = blockIdx.x, tid = threadIdx.x;
    __shared__ int s[128];
    s[tid] = sampled[b * 128 + tid];
    __syncthreads();
    bitonic128(s, tid);
    int dd = (tid > 0 && s[tid] == s[tid - 1]) ? 0 : s[tid];
    __syncthreads();
    s[tid] = dd;
    __syncthreads();
    bitonic128(s, tid);
    if (tid == 0) { ids_int[b * 129] = 0; out_ids[b * 129] = 0.f; out_mask[b * 129] = 1.f; }
    ids_int[b * 129 + 1 + tid]  = s[tid];
    out_ids[b * 129 + 1 + tid]  = (float)s[tid];
    out_mask[b * 129 + 1 + tid] = (s[tid] != 0) ? 1.f : 0.f;
}

// ---------------------------------------------------------------------------
// pack per (b,h): Qg(144x64 pre-scaled), Kb(288x64 = BT for scores GEMM),
// VT(64x288 = BT for A*V GEMM), all bf16 zero-padded.
// ---------------------------------------------------------------------------
__global__ __launch_bounds__(256)
void pack_kernel(const float* __restrict__ qkv, const int* __restrict__ ids_int,
                 bf16_t* __restrict__ Qg, bf16_t* __restrict__ Kb, bf16_t* __restrict__ VT)
{
    int bh = blockIdx.x, b = bh >> 4, h = bh & 15, tid = threadIdx.x;
    const long long base = (long long)b * 257 * 3072 + h * 64;

    bf16_t* q = Qg + (long long)bh * 144 * 64;
    for (int idx = tid; idx < 144 * 64; idx += 256) {
        int i = idx >> 6, d = idx & 63;
        float v = 0.f;
        if (i < 129) { int tok = ids_int[b * 129 + i]; v = 0.125f * qkv[base + (long long)tok * 3072 + d]; }
        q[idx] = (bf16_t)v;
    }
    bf16_t* kb = Kb + (long long)bh * 288 * 64;            // [j][d]
    for (int idx = tid; idx < 288 * 64; idx += 256) {
        int j = idx >> 6, d = idx & 63;
        float v = (j < 257) ? qkv[base + (long long)j * 3072 + 1024 + d] : 0.f;
        kb[idx] = (bf16_t)v;
    }
    bf16_t* vt = VT + (long long)bh * 64 * 288;            // [d][j]
    for (int idx = tid; idx < 64 * 288; idx += 256) {
        int d = idx / 288, j = idx % 288;
        float v = (j < 257) ? qkv[base + (long long)j * 3072 + 2048 + d] : 0.f;
        vt[idx] = (bf16_t)v;
    }
}

// ---------------------------------------------------------------------------
__global__ __launch_bounds__(256)
void softmax_kernel(const float* __restrict__ S, bf16_t* __restrict__ Attn)
{
    int blk = blockIdx.x, r = blk % 144, bh = blk / 144, tid = threadIdx.x;
    const float* srow = S    + ((long long)bh * 144 + r) * 288;
    bf16_t*      arow = Attn + ((long long)bh * 144 + r) * 288;
    if (r >= 129) { for (int j = tid; j < 288; j += 256) arow[j] = (bf16_t)0.f; return; }

    __shared__ float red[256];
    float x0 = (tid < 257) ? srow[tid] : NEG_INF;
    float x1 = (tid + 256 < 257) ? srow[tid + 256] : NEG_INF;
    red[tid] = fmaxf(x0, x1); __syncthreads();
    for (int s = 128; s > 0; s >>= 1) { if (tid < s) red[tid] = fmaxf(red[tid], red[tid + s]); __syncthreads(); }
    float m = red[0]; __syncthreads();

    float p0 = (tid < 257) ? expf(x0 - m) : 0.f;
    float p1 = (tid + 256 < 257) ? expf(x1 - m) : 0.f;
    red[tid] = p0 + p1; __syncthreads();
    for (int s = 128; s > 0; s >>= 1) { if (tid < s) red[tid] += red[tid + s]; __syncthreads(); }
    float denom = red[0];

    arow[tid] = (bf16_t)((tid < 257) ? (p0 / denom) : 0.f);
    int j2 = tid + 256;
    if (j2 < 288) arow[j2] = (bf16_t)((j2 < 257) ? (p1 / denom) : 0.f);
}

// ---------------------------------------------------------------------------
__global__ void reorder_kernel(const float* __restrict__ O, bf16_t* __restrict__ Og)
{
    int row = blockIdx.x;                 // 0..2063
    int b = row / 129, i = row % 129, tid = threadIdx.x;
    for (int c = tid; c < 1024; c += 256) {
        int h = c >> 6, d = c & 63;
        Og[(long long)row * 1024 + c] =
            (bf16_t)O[(((long long)(b * 16 + h) * 144) + i) * 64 + d];
    }
}

// ---------------------------------------------------------------------------
extern "C" void kernel_launch(void* const* d_in, const int* in_sizes, int n_in,
                              void* d_out, int out_size, void* d_ws, size_t ws_size,
                              hipStream_t stream)
{
    (void)in_sizes; (void)n_in; (void)out_size; (void)ws_size;
    const float*         x    = (const float*)d_in[0];
    const unsigned char* mask = (const unsigned char*)d_in[1];
    const float*         gu   = (const float*)d_in[2];
    const float*         Wqkv = (const float*)d_in[3];
    const float*         Wout = (const float*)d_in[4];
    const float*         bout = (const float*)d_in[5];

    const int B = 16, DIM = 1024, KT_TOK = 128;
    const int BN = 16 * 257;                    // 4112
    const int INNER3 = 3072;
    const int MR = B * (KT_TOK + 1);            // 2064 output rows

    // ---- workspace layout ----
    size_t off = 0;
    auto alloc = [&](size_t bytes) { size_t o = off; off = (off + bytes + 255) & ~(size_t)255; return o; };
    char* ws = (char*)d_ws;
    size_t o_xbf   = alloc((size_t)BN * DIM * 2);
    size_t o_wqkv  = alloc((size_t)DIM * INNER3 * 2);      // transposed [3072][1024]
    size_t o_wout  = alloc((size_t)DIM * DIM * 2);         // transposed [1024][1024]
    size_t o_qkv   = alloc((size_t)BN * INNER3 * 4);       // f32; later aliased by S
    size_t o_cls   = alloc((size_t)B * 256 * 4);
    size_t o_plog  = alloc((size_t)B * 256 * 4);
    size_t o_samp  = alloc((size_t)B * 128 * 4);
    size_t o_ids   = alloc((size_t)B * 129 * 4);
    size_t o_qg    = alloc((size_t)256 * 144 * 64 * 2);
    size_t o_kb    = alloc((size_t)256 * 288 * 64 * 2);
    size_t o_vt    = alloc((size_t)256 * 64 * 288 * 2);
    size_t o_attn  = alloc((size_t)256 * 144 * 288 * 2);
    size_t o_ofp   = alloc((size_t)256 * 144 * 64 * 4);
    size_t o_og    = alloc((size_t)MR * DIM * 2);

    bf16_t* x_bf   = (bf16_t*)(ws + o_xbf);
    bf16_t* wqkvT  = (bf16_t*)(ws + o_wqkv);
    bf16_t* woutT  = (bf16_t*)(ws + o_wout);
    float*  qkv    = (float*)(ws + o_qkv);
    float*  Sbuf   = qkv;                                  // alias (42.5MB <= 50.5MB, qkv dead)
    float*  cls    = (float*)(ws + o_cls);
    float*  plog   = (float*)(ws + o_plog);
    int*    samp   = (int*)(ws + o_samp);
    int*    idsi   = (int*)(ws + o_ids);
    bf16_t* Qg     = (bf16_t*)(ws + o_qg);
    bf16_t* Kb     = (bf16_t*)(ws + o_kb);
    bf16_t* VT     = (bf16_t*)(ws + o_vt);
    bf16_t* Attn   = (bf16_t*)(ws + o_attn);
    float*  Ofp    = (float*)(ws + o_ofp);
    bf16_t* Og     = (bf16_t*)(ws + o_og);

    float* out_f    = (float*)d_out;                       // 2064x1024
    float* out_mask = out_f + (size_t)MR * DIM;            // 2064
    float* out_ids  = out_mask + MR;                       // 2064

    // 1. converts (weights transposed for BT-layout GEMM)
    long long nx = (long long)BN * DIM;
    f2bf_kernel<<<dim3((nx + 255) / 256), 256, 0, stream>>>(x, x_bf, nx);
    f2bf_T_kernel<<<dim3(INNER3 / 32, DIM / 32), 256, 0, stream>>>(Wqkv, wqkvT, DIM, INNER3);
    f2bf_T_kernel<<<dim3(DIM / 32, DIM / 32), 256, 0, stream>>>(Wout, woutT, DIM, DIM);

    // 2. QKV GEMM: (4112x1024) x (1024x3072)
    gemm_bf16_wmma<<<dim3((INNER3 + 127) / 128, (BN + 127) / 128, 1), 256, 0, stream>>>(
        x_bf, wqkvT, qkv, nullptr, BN, INNER3, DIM, 0, 0, 0);

    // 3. cls scoring
    zero_f32_kernel<<<dim3(16), 256, 0, stream>>>(cls, (long long)B * 256);
    score_kernel<<<dim3(B * 16), 256, 0, stream>>>(qkv, cls);

    // 4. sampling pipeline
    logits_kernel<<<dim3(B), 256, 0, stream>>>(cls, mask, plog);
    gumbel_argmax_kernel<<<dim3(B * KT_TOK), 256, 0, stream>>>(plog, gu, samp);
    ids_kernel<<<dim3(B), 128, 0, stream>>>(samp, idsi, out_mask, out_ids);

    // 5. pack gathered Q, K (BT layout), V^T (BT layout)
    pack_kernel<<<dim3(B * 16), 256, 0, stream>>>(qkv, idsi, Qg, Kb, VT);

    // 6. scores: batched (144x64) x (64x288), BT = Kb(288x64), z = 256
    gemm_bf16_wmma<<<dim3(3, 2, 256), 256, 0, stream>>>(
        Qg, Kb, Sbuf, nullptr, 144, 288, 64,
        (long long)144 * 64, (long long)288 * 64, (long long)144 * 288);

    // 7. softmax -> bf16 attn
    softmax_kernel<<<dim3(256 * 144), 256, 0, stream>>>(Sbuf, Attn);

    // 8. A*V: batched (144x288) x (288x64), BT = VT(64x288)
    gemm_bf16_wmma<<<dim3(1, 2, 256), 256, 0, stream>>>(
        Attn, VT, Ofp, nullptr, 144, 64, 288,
        (long long)144 * 288, (long long)64 * 288, (long long)144 * 64);

    // 9. reorder heads -> (2064 x 1024) bf16
    reorder_kernel<<<dim3(MR), 256, 0, stream>>>(Ofp, Og);

    // 10. output projection with bias: (2064x1024) x (1024x1024) -> d_out
    gemm_bf16_wmma<<<dim3(DIM / 128, (MR + 127) / 128, 1), 256, 0, stream>>>(
        Og, woutT, out_f, bout, MR, DIM, DIM, 0, 0, 0);
}
